// MultiHeadAttention_43044162240991
// MI455X (gfx1250) — matrix operand order repaired and checked by reading, not compiled, not run on hardware
//
#include <hip/hip_runtime.h>

// ---------------- types ----------------
typedef __attribute__((ext_vector_type(16))) __bf16 v16bf;
typedef __attribute__((ext_vector_type(8)))  __bf16 v8bf;
typedef __attribute__((ext_vector_type(8)))  float  v8f;

#define S_LEN  2048
#define DMODEL 512
#define NHEADS 8
#define HDIM   64
#define BATCH  4
#define MROWS  (BATCH * S_LEN)   // 8192

// ---------------- WMMA wrapper ----------------
__device__ __forceinline__ v8f wmma_bf16(v16bf a, v16bf b, v8f c) {
  return __builtin_amdgcn_wmma_f32_16x16x32_bf16(false, a, false, b, (short)0, c,
                                                 false, false);
}

// ---------------- CDNA5 async copy (global -> LDS), ASYNCcnt tracked ---------
__device__ __forceinline__ void async_copy_b128(void* lds_ptr, const void* gptr) {
  // VDST = LDS byte offset (low 32 bits of flat shared address), VADDR = 64-bit
  // global address. Tracked with ASYNCcnt.
  unsigned lds_off = (unsigned)(uintptr_t)lds_ptr;
  asm volatile("global_load_async_to_lds_b128 %0, %1, off"
               :
               : "v"(lds_off), "v"(gptr)
               : "memory");
}
__device__ __forceinline__ void wait_async0() {
  asm volatile("s_wait_asynccnt 0x0" ::: "memory");
}
__device__ __forceinline__ void wait_async1() {
  asm volatile("s_wait_asynccnt 0x1" ::: "memory");
}
__device__ __forceinline__ void wait_async2() {
  asm volatile("s_wait_asynccnt 0x2" ::: "memory");
}

// A fragment: 16x32 (MxK) bf16, row-major [row, k], leading dim = ld.
// Lane L: m=L%16, half=L/16. e<8 -> k=k0+8*half+e ; e>=8 -> k=k0+16+8*half+(e-8)
__device__ __forceinline__ v16bf load_a(const __bf16* A, int ld, int row, int k0,
                                        int half) {
  const __bf16* p = A + (size_t)row * ld + k0 + 8 * half;
  v8bf lo = *(const v8bf*)(p);
  v8bf hi = *(const v8bf*)(p + 16);
  v16bf f;
#pragma unroll
  for (int i = 0; i < 8; ++i) { f[i] = lo[i]; f[8 + i] = hi[i]; }
  return f;
}

// B fragment: 32x16 (KxN) bf16, K-major storage: Bt[n*ld + k].
// Lane L: n = n0 + L%16, half = L/16. e<8 -> k=k0+8*half+e ; e>=8 -> k=k0+16+8*half+(e-8)
__device__ __forceinline__ v16bf load_b(const __bf16* Bt, int ld, int n0, int k0,
                                        int ln, int half) {
  const __bf16* p = Bt + (size_t)(n0 + ln) * ld + k0 + 8 * half;
  v8bf lo = *(const v8bf*)(p);
  v8bf hi = *(const v8bf*)(p + 16);
  v16bf f;
#pragma unroll
  for (int i = 0; i < 8; ++i) { f[i] = lo[i]; f[8 + i] = hi[i]; }
  return f;
}

// ---------------- conversion kernels ----------------
__global__ void convert_x_kernel(const float* __restrict__ x, void* Xb_raw, int n) {
  __bf16* Xb = (__bf16*)Xb_raw;
  int i = blockIdx.x * blockDim.x + threadIdx.x;
  if (i < n) Xb[i] = (__bf16)x[i];
}

// Transpose + convert: Wt[n*512 + k] = W[k*512 + n]
__global__ void convert_w_kernel(const float* __restrict__ Wq,
                                 const float* __restrict__ Wk,
                                 const float* __restrict__ Wv,
                                 const float* __restrict__ Wo,
                                 void* WqT_raw, void* WkT_raw,
                                 void* WvT_raw, void* WoT_raw) {
  int i = blockIdx.x * blockDim.x + threadIdx.x;   // 0 .. 512*512-1
  int z = blockIdx.y;
  const float* src = (z == 0) ? Wq : (z == 1) ? Wk : (z == 2) ? Wv : Wo;
  __bf16* dst = (__bf16*)((z == 0) ? WqT_raw : (z == 1) ? WkT_raw
                                    : (z == 2) ? WvT_raw : WoT_raw);
  int k = i >> 9;
  int n = i & 511;
  dst[(size_t)n * DMODEL + k] = (__bf16)src[i];
}

// ---------------- QKV projection GEMM ----------------
// grid = (M/128, N/64, 3); block = 256 (8 waves). Wave owns 16(M) x 64(N).
// Weight tile (64n x 32k) staged per block via async-LDS, double buffered.
__global__ __launch_bounds__(256) void qkv_gemm_kernel(
    const void* Xb_raw, const void* WqT_raw, const void* WkT_raw,
    const void* WvT_raw, const float* __restrict__ bq,
    const float* __restrict__ bk, const float* __restrict__ bv,
    void* Qb_raw, void* Kb_raw, void* Vt_raw) {
  __shared__ __bf16 sB[2][64 * 32];   // [n][k], 4 KB per stage

  const __bf16* Xb = (const __bf16*)Xb_raw;
  const int tid = threadIdx.x;
  const int lane = tid & 31;
  const int wave = tid >> 5;
  const int ln = lane & 15, half = lane >> 4;
  const int z = blockIdx.z;

  const __bf16* WT = (const __bf16*)((z == 0) ? WqT_raw
                                     : (z == 1) ? WkT_raw : WvT_raw);
  const float* bias = (z == 0) ? bq : (z == 1) ? bk : bv;

  const int mbase = blockIdx.x * 128 + wave * 16;
  const int nbase = blockIdx.y * 64;
  const int row = mbase + ln;

  // async-staging assignment: thread -> (n row, 16B chunk)
  const int cn = tid >> 2;          // 0..63
  const int ck = (tid & 3) * 8;     // element offset 0/8/16/24

  v8f acc[4];
#pragma unroll
  for (int t = 0; t < 4; ++t) acc[t] = (v8f){0, 0, 0, 0, 0, 0, 0, 0};

  const int NSTEP = DMODEL / 32;    // 16
  // stage 0
  async_copy_b128(&sB[0][cn * 32 + ck],
                  WT + (size_t)(nbase + cn) * DMODEL + 0 + ck);

  v16bf a_cur = load_a(Xb, DMODEL, row, 0, half);

  for (int s = 0; s < NSTEP; ++s) {
    if (s + 1 < NSTEP) {
      async_copy_b128(&sB[(s + 1) & 1][cn * 32 + ck],
                      WT + (size_t)(nbase + cn) * DMODEL + (s + 1) * 32 + ck);
      wait_async1();               // stage s complete (in-order)
    } else {
      wait_async0();
    }
    __syncthreads();

    v16bf a_next;
    if (s + 1 < NSTEP) a_next = load_a(Xb, DMODEL, row, (s + 1) * 32, half);

    const __bf16* bb = &sB[s & 1][0];
#pragma unroll
    for (int t = 0; t < 4; ++t) {
      v16bf bfr = load_b(bb, 32, t * 16, 0, ln, half);   // ds_load path
      acc[t] = wmma_bf16(a_cur, bfr, acc[t]);
    }
    a_cur = a_next;
    __syncthreads();               // all waves done reading before overwrite
  }

  __bf16* Qb = (__bf16*)Qb_raw;
  __bf16* Kb = (__bf16*)Kb_raw;
  __bf16* Vt = (__bf16*)Vt_raw;
#pragma unroll
  for (int t = 0; t < 4; ++t) {
    int n = nbase + t * 16 + ln;
    float bval = bias[n];
    int h = n >> 6, hd = n & 63;
#pragma unroll
    for (int v = 0; v < 8; ++v) {
      int r = mbase + v + 8 * half;
      int bb = r >> 11;              // / S_LEN
      int sdx = r & (S_LEN - 1);
      float val = acc[t][v] + bval;
      if (z == 0)
        Qb[(((size_t)bb * NHEADS + h) * S_LEN + sdx) * HDIM + hd] = (__bf16)val;
      else if (z == 1)
        Kb[(((size_t)bb * NHEADS + h) * S_LEN + sdx) * HDIM + hd] = (__bf16)val;
      else
        Vt[(((size_t)bb * NHEADS + h) * HDIM + hd) * S_LEN + sdx] = (__bf16)val;
    }
  }
}

// ---------------- flash attention ----------------
// grid = (S/128, B*H); block = 256 (8 waves). Wave owns 16 query rows.
// K chunk (32x64) and V chunk (64x32) staged via async-LDS, double buffered.
__global__ __launch_bounds__(256) void attn_kernel(
    const void* Qb_raw, const void* Kb_raw, const void* Vt_raw,
    const int* __restrict__ mask, void* Ab_raw) {
  __shared__ __bf16 kshm[2][32 * 64];   // [kpos][hd]
  __shared__ __bf16 vshm[2][64 * 32];   // [hd][kpos]
  __shared__ __bf16 pshm[8][16 * 32];   // per-wave P tile

  const __bf16* Qb = (const __bf16*)Qb_raw;
  const __bf16* Kb = (const __bf16*)Kb_raw;
  const __bf16* Vt = (const __bf16*)Vt_raw;
  __bf16* Ab = (__bf16*)Ab_raw;

  const int tid = threadIdx.x;
  const int lane = tid & 31;
  const int wave = tid >> 5;
  const int ln = lane & 15, half = lane >> 4;
  const int bh = blockIdx.y;           // 0..31
  const int b = bh >> 3, h = bh & 7;

  const __bf16* Qh = Qb + (size_t)bh * S_LEN * HDIM;
  const __bf16* Kh = Kb + (size_t)bh * S_LEN * HDIM;
  const __bf16* Vh = Vt + (size_t)bh * HDIM * S_LEN;

  const int qbase = blockIdx.x * 128 + wave * 16;
  const int qrow = qbase + ln;

  // staging assignment
  const int kr = tid >> 3, kcoff = (tid & 7) * 8;   // K: 32 rows x 8 chunks16B
  const int vr = tid >> 2, vcoff = (tid & 3) * 8;   // V: 64 rows x 4 chunks16B

  // Q A-fragments preloaded once (K-dim = head dim, 2 steps of 32)
  v16bf aq0 = load_a(Qh, HDIM, qrow, 0, half);
  v16bf aq1 = load_a(Qh, HDIM, qrow, 32, half);

  v8f o[4];
#pragma unroll
  for (int t = 0; t < 4; ++t) o[t] = (v8f){0, 0, 0, 0, 0, 0, 0, 0};
  float m_i[8], l_i[8];
#pragma unroll
  for (int v = 0; v < 8; ++v) { m_i[v] = -1e30f; l_i[v] = 0.f; }

  __bf16* pb = &pshm[wave][0];

  const int NCH = S_LEN / 32;    // 64 chunks
  // stage 0
  async_copy_b128(&kshm[0][kr * 64 + kcoff], Kh + (size_t)kr * HDIM + kcoff);
  async_copy_b128(&vshm[0][vr * 32 + vcoff], Vh + (size_t)vr * S_LEN + vcoff);

  for (int c = 0; c < NCH; ++c) {
    const int kc = c * 32;
    if (c + 1 < NCH) {
      const int kn = kc + 32;
      async_copy_b128(&kshm[(c + 1) & 1][kr * 64 + kcoff],
                      Kh + (size_t)(kn + kr) * HDIM + kcoff);
      async_copy_b128(&vshm[(c + 1) & 1][vr * 32 + vcoff],
                      Vh + (size_t)vr * S_LEN + kn + vcoff);
      wait_async2();              // stage c (2 ops) complete
    } else {
      wait_async0();
    }
    __syncthreads();

    const __bf16* kb = &kshm[c & 1][0];
    const __bf16* vb = &vshm[c & 1][0];

    v8f s0 = (v8f){0, 0, 0, 0, 0, 0, 0, 0};
    v8f s1 = (v8f){0, 0, 0, 0, 0, 0, 0, 0};
    // scores: Q(16x64) x K^T(64x32) -> two 16x16 tiles (B-frags from LDS)
    s0 = wmma_bf16(aq0, load_b(kb, HDIM, 0, 0, ln, half), s0);
    s0 = wmma_bf16(aq1, load_b(kb, HDIM, 0, 32, ln, half), s0);
    s1 = wmma_bf16(aq0, load_b(kb, HDIM, 16, 0, ln, half), s1);
    s1 = wmma_bf16(aq1, load_b(kb, HDIM, 16, 32, ln, half), s1);

    float mb0 = (mask[b * S_LEN + kc + ln] == 0) ? -1e9f : 0.f;
    float mb1 = (mask[b * S_LEN + kc + 16 + ln] == 0) ? -1e9f : 0.f;

    float cm[8];
#pragma unroll
    for (int v = 0; v < 8; ++v) {
      s0[v] = s0[v] * 0.125f + mb0;    // / sqrt(64)
      s1[v] = s1[v] * 0.125f + mb1;
      cm[v] = fmaxf(s0[v], s1[v]);
    }
#pragma unroll
    for (int off = 1; off < 16; off <<= 1)
#pragma unroll
      for (int v = 0; v < 8; ++v)
        cm[v] = fmaxf(cm[v], __shfl_xor(cm[v], off, 32));

    float rs[8];
#pragma unroll
    for (int v = 0; v < 8; ++v) {
      float mn = fmaxf(m_i[v], cm[v]);
      float alpha = __expf(m_i[v] - mn);
      m_i[v] = mn;
      l_i[v] *= alpha;
      float p0 = __expf(s0[v] - mn);
      float p1 = __expf(s1[v] - mn);
      s0[v] = p0;
      s1[v] = p1;
      rs[v] = p0 + p1;
#pragma unroll
      for (int t = 0; t < 4; ++t) o[t][v] *= alpha;
    }
#pragma unroll
    for (int off = 1; off < 16; off <<= 1)
#pragma unroll
      for (int v = 0; v < 8; ++v) rs[v] += __shfl_xor(rs[v], off, 32);
#pragma unroll
    for (int v = 0; v < 8; ++v) l_i[v] += rs[v];

    // re-layout P: C-layout -> per-wave LDS tile -> A-fragment
#pragma unroll
    for (int v = 0; v < 8; ++v) {
      int r = v + 8 * half;
      pb[r * 32 + ln] = (__bf16)s0[v];
      pb[r * 32 + 16 + ln] = (__bf16)s1[v];
    }
    asm volatile("s_wait_dscnt 0x0" ::: "memory");

    v16bf ap;
    {
      const __bf16* pr = pb + ln * 32 + 8 * half;
      v8bf lo = *(const v8bf*)(pr);
      v8bf hi = *(const v8bf*)(pr + 16);
#pragma unroll
      for (int i = 0; i < 8; ++i) { ap[i] = lo[i]; ap[8 + i] = hi[i]; }
    }

    // out += P(16x32) x V(32x64): 4 N-tiles over head dim (B-frags from LDS)
#pragma unroll
    for (int t = 0; t < 4; ++t) {
      v16bf bv = load_b(vb, 32, t * 16, 0, ln, half);
      o[t] = wmma_bf16(ap, bv, o[t]);
    }
    __syncthreads();               // all waves done reading before overwrite
  }

  // epilogue: divide by softmax denominator, write [B*S, D] bf16
#pragma unroll
  for (int t = 0; t < 4; ++t) {
    int hd = t * 16 + ln;
#pragma unroll
    for (int v = 0; v < 8; ++v) {
      int q = qbase + v + 8 * half;
      float val = o[t][v] / l_i[v];
      Ab[((size_t)(b * S_LEN + q)) * DMODEL + h * HDIM + hd] = (__bf16)val;
    }
  }
}

// ---------------- output projection GEMM ----------------
// grid = (M/128, N/64); f32 output directly into d_out [B,S,D]
__global__ __launch_bounds__(256) void out_gemm_kernel(
    const void* Ab_raw, const void* WoT_raw, const float* __restrict__ bo,
    float* __restrict__ out) {
  __shared__ __bf16 sB[2][64 * 32];

  const __bf16* Ab = (const __bf16*)Ab_raw;
  const __bf16* WoT = (const __bf16*)WoT_raw;
  const int tid = threadIdx.x;
  const int lane = tid & 31;
  const int wave = tid >> 5;
  const int ln = lane & 15, half = lane >> 4;

  const int mbase = blockIdx.x * 128 + wave * 16;
  const int nbase = blockIdx.y * 64;
  const int row = mbase + ln;

  const int cn = tid >> 2;
  const int ck = (tid & 3) * 8;

  v8f acc[4];
#pragma unroll
  for (int t = 0; t < 4; ++t) acc[t] = (v8f){0, 0, 0, 0, 0, 0, 0, 0};

  const int NSTEP = DMODEL / 32;
  async_copy_b128(&sB[0][cn * 32 + ck],
                  WoT + (size_t)(nbase + cn) * DMODEL + 0 + ck);
  v16bf a_cur = load_a(Ab, DMODEL, row, 0, half);

  for (int s = 0; s < NSTEP; ++s) {
    if (s + 1 < NSTEP) {
      async_copy_b128(&sB[(s + 1) & 1][cn * 32 + ck],
                      WoT + (size_t)(nbase + cn) * DMODEL + (s + 1) * 32 + ck);
      wait_async1();
    } else {
      wait_async0();
    }
    __syncthreads();

    v16bf a_next;
    if (s + 1 < NSTEP) a_next = load_a(Ab, DMODEL, row, (s + 1) * 32, half);

    const __bf16* bb = &sB[s & 1][0];
#pragma unroll
    for (int t = 0; t < 4; ++t) {
      v16bf bfr = load_b(bb, 32, t * 16, 0, ln, half);
      acc[t] = wmma_bf16(a_cur, bfr, acc[t]);
    }
    a_cur = a_next;
    __syncthreads();
  }

#pragma unroll
  for (int t = 0; t < 4; ++t) {
    int n = nbase + t * 16 + ln;
    float bval = bo[n];
#pragma unroll
    for (int v = 0; v < 8; ++v) {
      int r = mbase + v + 8 * half;
      out[(size_t)r * DMODEL + n] = acc[t][v] + bval;
    }
  }
}

// ---------------- launcher ----------------
extern "C" void kernel_launch(void* const* d_in, const int* in_sizes, int n_in,
                              void* d_out, int out_size, void* d_ws,
                              size_t ws_size, hipStream_t stream) {
  const float* x  = (const float*)d_in[0];
  const int* mask = (const int*)d_in[1];
  const float* Wq = (const float*)d_in[2];
  const float* bq = (const float*)d_in[3];
  const float* Wk = (const float*)d_in[4];
  const float* bk = (const float*)d_in[5];
  const float* Wv = (const float*)d_in[6];
  const float* bv = (const float*)d_in[7];
  const float* Wo = (const float*)d_in[8];
  const float* bo = (const float*)d_in[9];
  float* out = (float*)d_out;

  char* ws = (char*)d_ws;
  const size_t SZ_X = (size_t)MROWS * DMODEL * 2;    // 8 MB
  const size_t SZ_W = (size_t)DMODEL * DMODEL * 2;   // 0.5 MB
  size_t off = 0;
  void* Xb  = ws + off; off += SZ_X;
  void* WqT = ws + off; off += SZ_W;
  void* WkT = ws + off; off += SZ_W;
  void* WvT = ws + off; off += SZ_W;
  void* WoT = ws + off; off += SZ_W;
  void* Qb  = ws + off; off += SZ_X;   // [B,H,S,HD]
  void* Kb  = ws + off; off += SZ_X;   // [B,H,S,HD]
  void* Vt  = ws + off; off += SZ_X;   // [B,H,HD,S]
  void* Ab  = ws + off; off += SZ_X;   // [B*S, D]
  (void)ws_size; (void)in_sizes; (void)n_in; (void)out_size;

  dim3 blk(256);
  convert_x_kernel<<<(MROWS * DMODEL + 255) / 256, blk, 0, stream>>>(
      x, Xb, MROWS * DMODEL);
  convert_w_kernel<<<dim3(DMODEL * DMODEL / 256, 4), blk, 0, stream>>>(
      Wq, Wk, Wv, Wo, WqT, WkT, WvT, WoT);
  qkv_gemm_kernel<<<dim3(MROWS / 128, DMODEL / 64, 3), blk, 0, stream>>>(
      Xb, WqT, WkT, WvT, bq, bk, bv, Qb, Kb, Vt);
  attn_kernel<<<dim3(S_LEN / 128, BATCH * NHEADS), blk, 0, stream>>>(
      Qb, Kb, Vt, mask, Ab);
  out_gemm_kernel<<<dim3(MROWS / 128, DMODEL / 64), blk, 0, stream>>>(
      Ab, WoT, bo, out);
}
